// Net1_83811991814568
// MI455X (gfx1250) — matrix operand order repaired and checked by reading, not compiled, not run on hardware
//
#include <hip/hip_runtime.h>
#include <math.h>

// ---------------------------------------------------------------------------
// Types for CDNA5 WMMA
// ---------------------------------------------------------------------------
typedef __attribute__((ext_vector_type(16))) __bf16 v16bf;
typedef __attribute__((ext_vector_type(8)))  float  v8f;

#define ACT_NONE 0
#define ACT_RELU 1
#define ACT_TANH 2
#define ACT_SIG  3

// ---------------------------------------------------------------------------
// General GEMM: C[M,N] = act(A[M,K] @ B[K,N]), fp32 in/out, bf16 WMMA compute
// Block tile 64x128, 256 threads = 8 waves, wave tile 32x32 (2x2 WMMA 16x16),
// K-step 32.  Software-pipelined: next K-tile's global loads issue between the
// LDS store and the WMMA compute of the current tile.  Full K-tiles need no
// per-element guards (OOB rows/cols are clamped and only feed C entries the
// guarded store never writes); the K-tail tile zero-masks.  When lda is a
// multiple of 4 the A staging uses b128 loads.
// ---------------------------------------------------------------------------
#define BM 64
#define BN 128
#define BK 32
#define SA 40   // padded LDS row stride (halves) for A   (80 B, rows 16B-aligned)
#define SB 40   // padded LDS row stride (halves) for B^T (80 B, rows 16B-aligned)

__device__ __forceinline__ int clampi(int x, int hi) { return x < hi ? x : hi; }

__launch_bounds__(256, 2)
__global__ void gemm_bf16_kernel(const float* __restrict__ A,
                                 const float* __restrict__ B,
                                 float* __restrict__ C,
                                 int M, int N, int K,
                                 int lda, int ldb, int ldc, int act)
{
    __shared__ __align__(16) __bf16 sA[BM * SA];
    __shared__ __align__(16) __bf16 sB[BN * SB];   // transposed: [n][k]

    const int tid  = threadIdx.x;
    const int wave = tid >> 5;
    const int lane = tid & 31;
    const int half = lane >> 4;
    const int l15  = lane & 15;

    const int m0 = blockIdx.y * BM;
    const int n0 = blockIdx.x * BN;
    const int waveM = (wave >> 2) * 32;   // 0 or 32
    const int waveN = (wave & 3) * 32;    // 0,32,64,96

    // staging coordinates (base row/col clamped once; no per-element guards
    // in full tiles -- clamped data only reaches never-stored C entries)
    const int arow = tid >> 2;            // 0..63   (A: row in tile)
    const int akq  = (tid & 3) * 8;       // 0,8,16,24 (A: 8 contiguous k)
    const int agmc = clampi(m0 + arow, M - 1);
    const float* __restrict__ Abase = A + (long)agmc * lda;
    const bool aVec = ((lda & 3) == 0);   // rows 16B-aligned -> b128 loads

    const int bn   = tid & 127;           // B: column in tile
    const int bkh  = (tid >> 7) * 16;     // 0 or 16 (B: 16 contiguous k)
    const int bgnc = clampi(n0 + bn, N - 1);
    const float* __restrict__ Bbase = B + bgnc;

    v8f c00 = {}, c01 = {}, c10 = {}, c11 = {};

    const int nk = (K + BK - 1) / BK;

    float fA[8];
    float fB[16];
    auto loadTile = [&](int kt) {
        const int k0 = kt * BK;
        if (k0 + BK <= K) {          // uniform branch: full tile, no guards
            const float* pa = Abase + k0 + akq;
            if (aVec) {              // uniform: 2x global_load_b128
                const float4 u = *(const float4*)(pa);
                const float4 v = *(const float4*)(pa + 4);
                fA[0] = u.x; fA[1] = u.y; fA[2] = u.z; fA[3] = u.w;
                fA[4] = v.x; fA[5] = v.y; fA[6] = v.z; fA[7] = v.w;
            } else {
                #pragma unroll
                for (int j = 0; j < 8; ++j) fA[j] = pa[j];
            }
            const float* pb = Bbase + (long)(k0 + bkh) * ldb;
            #pragma unroll
            for (int j = 0; j < 16; ++j) fB[j] = pb[(long)j * ldb];
        } else {                     // K-tail: zero-mask out-of-range k
            #pragma unroll
            for (int j = 0; j < 8; ++j) {
                const int gk = k0 + akq + j;
                const float m = (gk < K) ? 1.0f : 0.0f;
                fA[j] = Abase[clampi(gk, K - 1)] * m;
            }
            #pragma unroll
            for (int j = 0; j < 16; ++j) {
                const int gk = k0 + bkh + j;
                const float m = (gk < K) ? 1.0f : 0.0f;
                fB[j] = Bbase[(long)clampi(gk, K - 1) * ldb] * m;
            }
        }
    };
    auto storeTileLDS = [&]() {
        union { __bf16 h[8]; uint4 q; } ka;
        #pragma unroll
        for (int j = 0; j < 8; ++j) ka.h[j] = (__bf16)fA[j];
        *(uint4*)&sA[arow * SA + akq] = ka.q;
        union { __bf16 h[16]; uint4 q[2]; } kb;
        #pragma unroll
        for (int j = 0; j < 16; ++j) kb.h[j] = (__bf16)fB[j];
        *(uint4*)&sB[bn * SB + bkh]     = kb.q[0];
        *(uint4*)&sB[bn * SB + bkh + 8] = kb.q[1];
    };

    loadTile(0);
    for (int kt = 0; kt < nk; ++kt) {
        storeTileLDS();
        __syncthreads();

        // issue next tile's global loads now; they overlap the DS reads and
        // WMMAs below and are only waited on at the next storeTileLDS()
        if (kt + 1 < nk) loadTile(kt + 1);

        // ---- build fragments (explicit b128 LDS reads) ----
        // A 16-bit layout: lanes 0-15 hold K {0..7, 16..23}; lanes 16-31
        // hold K {8..15, 24..31} for row M = lane&15.
        union Frag { v16bf v; uint4 q[2]; };
        Frag a0, a1, b0, b1;
        {
            const __bf16* pa0 = &sA[(waveM + l15) * SA + half * 8];
            const __bf16* pa1 = &sA[(waveM + 16 + l15) * SA + half * 8];
            a0.q[0] = *(const uint4*)(pa0);
            a0.q[1] = *(const uint4*)(pa0 + 16);
            a1.q[0] = *(const uint4*)(pa1);
            a1.q[1] = *(const uint4*)(pa1 + 16);
            // B 16-bit layout: lanes 0-15 hold K {0..7, 16..23} of column n;
            // lanes 16-31 hold K {8..15, 24..31}.  sB is stored [n][k].
            const __bf16* pb0 = &sB[(waveN + l15) * SB + half * 8];
            const __bf16* pb1 = &sB[(waveN + 16 + l15) * SB + half * 8];
            b0.q[0] = *(const uint4*)(pb0);
            b0.q[1] = *(const uint4*)(pb0 + 16);
            b1.q[0] = *(const uint4*)(pb1);
            b1.q[1] = *(const uint4*)(pb1 + 16);
        }

        c00 = __builtin_amdgcn_wmma_f32_16x16x32_bf16(false, a0.v, false, b0.v, (short)0, c00, false, false);
        c01 = __builtin_amdgcn_wmma_f32_16x16x32_bf16(false, a0.v, false, b1.v, (short)0, c01, false, false);
        c10 = __builtin_amdgcn_wmma_f32_16x16x32_bf16(false, a1.v, false, b0.v, (short)0, c10, false, false);
        c11 = __builtin_amdgcn_wmma_f32_16x16x32_bf16(false, a1.v, false, b1.v, (short)0, c11, false, false);
        __syncthreads();
    }

    // ---- store with fused activation ----
    // C layout: VGPR r, lane L -> M = 8*(L>=16) + r, N = L&15
    auto actf = [&](float x) -> float {
        if (act == ACT_RELU) return x > 0.0f ? x : 0.0f;
        if (act == ACT_TANH) return tanhf(x);
        if (act == ACT_SIG)  return 1.0f / (1.0f + __expf(-x));
        return x;
    };
    auto store_tile = [&](const v8f& c, int tm, int tn) {
        const int n = n0 + tn + l15;
        if (n >= N) return;
        const int mbase = m0 + tm + half * 8;
        #pragma unroll
        for (int r = 0; r < 8; ++r) {
            const int m = mbase + r;
            if (m < M) C[(long)m * ldc + n] = actf(c[r]);
        }
    };
    store_tile(c00, waveM,      waveN);
    store_tile(c01, waveM,      waveN + 16);
    store_tile(c10, waveM + 16, waveN);
    store_tile(c11, waveM + 16, waveN + 16);
}

// ---------------------------------------------------------------------------
// Pooling: pool_feat[n,d] = mean_v h_v[n,d]; pool_view[n,v] = mean_d h_v[n,d]
// One 256-thread block per node.
// ---------------------------------------------------------------------------
__global__ void pool_kernel(const float* __restrict__ h0,
                            const float* __restrict__ h1,
                            const float* __restrict__ h2,
                            float* __restrict__ pool_feat,
                            float* __restrict__ pool_view)
{
    __shared__ float s[3][256];
    const int n = blockIdx.x;
    const int d = threadIdx.x;
    const long base = (long)n * 256 + d;
    const float a = h0[base], b = h1[base], c = h2[base];
    pool_feat[base] = (a + b + c) * (1.0f / 3.0f);
    s[0][d] = a; s[1][d] = b; s[2][d] = c;
    __syncthreads();
    for (int st = 128; st > 0; st >>= 1) {
        if (d < st) {
            s[0][d] += s[0][d + st];
            s[1][d] += s[1][d + st];
            s[2][d] += s[2][d + st];
        }
        __syncthreads();
    }
    if (d < 3) pool_view[(long)n * 3 + d] = s[d][0] * (1.0f / 256.0f);
}

// view_x = sigmoid(relu(pool_view @ vw1) @ vw2), tiny per-node op
__global__ void view_gate_kernel(const float* __restrict__ pool_view,
                                 const float* __restrict__ vw1,
                                 const float* __restrict__ vw2,
                                 float* __restrict__ view_x, int Nn)
{
    const int n = blockIdx.x * blockDim.x + threadIdx.x;
    if (n >= Nn) return;
    const float t0 = pool_view[(long)n * 3 + 0] * vw1[0] +
                     pool_view[(long)n * 3 + 1] * vw1[1] +
                     pool_view[(long)n * 3 + 2] * vw1[2];
    const float t = t0 > 0.0f ? t0 : 0.0f;
    #pragma unroll
    for (int v = 0; v < 3; ++v)
        view_x[(long)n * 3 + v] = 1.0f / (1.0f + __expf(-(t * vw2[v])));
}

// cat[n, :] = [view_x[n,0:3] | feat_x[n,0:256]]
__global__ void cat_kernel(const float* __restrict__ vx,
                           const float* __restrict__ fx,
                           float* __restrict__ cat, long total)
{
    const long i = (long)blockIdx.x * blockDim.x + threadIdx.x;
    if (i >= total) return;
    const long n = i / 259;
    const int  j = (int)(i % 259);
    cat[i] = (j < 3) ? vx[n * 3 + j] : fx[n * 256 + (j - 3)];
}

// z[n,:3] = pool_view * sigmoid(vf[:, :3]);  z[n,3:] = pool_feat * sigmoid(vf[:, 3:])
// (algebraic reduction of  h*view_w .mean(1)  and  h*feat_w .mean(2))
__global__ void zbuild_kernel(const float* __restrict__ vf,
                              const float* __restrict__ pv,
                              const float* __restrict__ pf,
                              float* __restrict__ z, long total)
{
    const long i = (long)blockIdx.x * blockDim.x + threadIdx.x;
    if (i >= total) return;
    const long n = i / 259;
    const int  j = (int)(i % 259);
    const float s = 1.0f / (1.0f + __expf(-vf[i]));
    z[i] = ((j < 3) ? pv[n * 3 + j] : pf[n * 256 + (j - 3)]) * s;
}

__global__ void zero_kernel(float* __restrict__ p, long total)
{
    const long i = (long)blockIdx.x * blockDim.x + threadIdx.x;
    if (i < total) p[i] = 0.0f;
}

// COO SpMM: out[row] += val * z[col].  One wave32 per edge, lanes over feats.
__global__ void spmm_kernel(const int* __restrict__ row,
                            const int* __restrict__ col,
                            const float* __restrict__ val,
                            const float* __restrict__ z,
                            float* __restrict__ out, int E)
{
    const int e = (int)(((long)blockIdx.x * blockDim.x + threadIdx.x) >> 5);
    const int lane = threadIdx.x & 31;
    if (e >= E) return;
    const int r = row[e];
    const int c = col[e];
    const float v = val[e];
    for (int f = lane; f < 259; f += 32)
        atomicAdd(&out[(long)r * 259 + f], v * z[(long)c * 259 + f]);
}

// ---------------------------------------------------------------------------
// Host orchestration
// ---------------------------------------------------------------------------
static inline void launch_gemm(const float* A, const float* B, float* C,
                               int M, int N, int K, int act, hipStream_t s)
{
    dim3 grid((N + BN - 1) / BN, (M + BM - 1) / BM);
    gemm_bf16_kernel<<<grid, 256, 0, s>>>(A, B, C, M, N, K, K, N, N, act);
}

extern "C" void kernel_launch(void* const* d_in, const int* in_sizes, int n_in,
                              void* d_out, int out_size, void* d_ws, size_t ws_size,
                              hipStream_t stream)
{
    (void)n_in; (void)out_size; (void)ws_size;
    const float* x0   = (const float*)d_in[0];
    const float* p0w1 = (const float*)d_in[1];
    const float* p0w2 = (const float*)d_in[2];
    const float* x1   = (const float*)d_in[3];
    const float* p1w1 = (const float*)d_in[4];
    const float* p1w2 = (const float*)d_in[5];
    const float* x2   = (const float*)d_in[6];
    const float* p2w1 = (const float*)d_in[7];
    const float* p2w2 = (const float*)d_in[8];
    const int*   erow = (const int*)d_in[9];
    const int*   ecol = (const int*)d_in[10];
    const float* evl  = (const float*)d_in[11];
    const float* vw1  = (const float*)d_in[12];
    const float* vw2  = (const float*)d_in[13];
    const float* fw1  = (const float*)d_in[14];
    const float* fw2  = (const float*)d_in[15];
    const float* m1   = (const float*)d_in[16];
    const float* m2   = (const float*)d_in[17];
    const float* g0   = (const float*)d_in[18];
    const float* g1   = (const float*)d_in[19];
    const float* g2   = (const float*)d_in[20];

    const int Nn = 50000;
    const int E  = in_sizes[9];

    // scratch carve-out
    size_t off = 0;
    auto carve = [&](size_t bytes) -> float* {
        float* p = (float*)((char*)d_ws + off);
        off += (bytes + 255) & ~(size_t)255;
        return p;
    };
    const size_t SZ256 = (size_t)Nn * 256 * sizeof(float);
    const size_t SZ259 = (size_t)Nn * 259 * sizeof(float);
    float* H0 = carve(SZ256);
    float* H1 = carve(SZ256);
    float* H2 = carve(SZ256);
    float* T  = carve(SZ256);                    // t / feat_x / vf1 (reused)
    float* PF = carve(SZ256);                    // pool_feat
    float* C1 = carve(SZ259);                    // cat -> z (ping)
    float* C2 = carve(SZ259);                    // vf  -> z (pong)
    float* C3 = carve(SZ259);                    // spmm accumulator
    float* PV = carve((size_t)Nn * 3  * sizeof(float));
    float* VX = carve((size_t)Nn * 3  * sizeof(float));
    float* FH = carve((size_t)Nn * 16 * sizeof(float));

    // ---- per-view projection MLPs ----
    launch_gemm(x0, p0w1, T,  Nn, 256, 1000, ACT_RELU, stream);
    launch_gemm(T,  p0w2, H0, Nn, 256, 256,  ACT_NONE, stream);
    launch_gemm(x1, p1w1, T,  Nn, 256, 800,  ACT_RELU, stream);
    launch_gemm(T,  p1w2, H1, Nn, 256, 256,  ACT_NONE, stream);
    launch_gemm(x2, p2w1, T,  Nn, 256, 600,  ACT_RELU, stream);
    launch_gemm(T,  p2w2, H2, Nn, 256, 256,  ACT_NONE, stream);

    // ---- pooling + gates ----
    pool_kernel<<<Nn, 256, 0, stream>>>(H0, H1, H2, PF, PV);
    launch_gemm(PF, fw1, FH, Nn, 16,  256, ACT_RELU, stream);   // hidden
    launch_gemm(FH, fw2, T,  Nn, 256, 16,  ACT_SIG,  stream);   // feat_x -> T
    view_gate_kernel<<<(Nn + 255) / 256, 256, 0, stream>>>(PV, vw1, vw2, VX, Nn);

    const long tot259 = (long)Nn * 259;
    const int  blk259 = (int)((tot259 + 255) / 256);
    cat_kernel<<<blk259, 256, 0, stream>>>(VX, T, C1, tot259);  // cat -> C1

    // ---- fusion MLP (m1/m2) ----
    launch_gemm(C1, m1, T,  Nn, 128, 259, ACT_RELU, stream);    // vf1 -> T
    launch_gemm(T,  m2, C2, Nn, 259, 128, ACT_NONE, stream);    // vf  -> C2
    zbuild_kernel<<<blk259, 256, 0, stream>>>(C2, PV, PF, C1, tot259);  // z -> C1

    // ---- 3-layer GCN: z = tanh(spmm(adj, z) @ g) ----
    float* zin  = C1;
    float* zout = C2;
    for (int l = 0; l < 3; ++l) {
        zero_kernel<<<blk259, 256, 0, stream>>>(C3, tot259);
        spmm_kernel<<<(E + 7) / 8, 256, 0, stream>>>(erow, ecol, evl, zin, C3, E);
        const float* g = (l == 0) ? g0 : (l == 1) ? g1 : g2;
        const int nc   = (l == 2) ? 10 : 259;
        float* dst     = (l == 2) ? (float*)d_out : zout;
        launch_gemm(C3, g, dst, Nn, nc, 259, ACT_TANH, stream);
        float* tmp = zin; zin = zout; zout = tmp;
    }
}
